// EncoderModule_4724464025770
// MI455X (gfx1250) — matrix operand order repaired and checked by reading, not compiled, not run on hardware
//
#include <hip/hip_runtime.h>
#include <hip/hip_bf16.h>
#include <math.h>

typedef __bf16 bf16;
typedef __attribute__((ext_vector_type(16))) __bf16 v16bf;
typedef __attribute__((ext_vector_type(8)))  __bf16 v8bf;
typedef __attribute__((ext_vector_type(8)))  float  v8f;

#define B_   4
#define S_   1024
#define H_   768
#define NH_  12
#define HD_  64
#define FF_  3072
#define MTOK (B_*S_)   // 4096
#define NQKV (3*H_)    // 2304

// ---------------------------------------------------------------------------
// CDNA5 primitives
// ---------------------------------------------------------------------------
__device__ __forceinline__ v8f wmma_bf16(v16bf a, v16bf b, v8f c) {
  // (neg_a, A, neg_b, B, c_mod, C, reuse_a, reuse_b)
  return __builtin_amdgcn_wmma_f32_16x16x32_bf16(false, a, false, b, (short)0, c, false, false);
}

// async copy 16B/lane: global -> LDS, tracked by ASYNCcnt (gfx1250)
__device__ __forceinline__ void async_cp16(unsigned lds_off, const void* gptr) {
  asm volatile("global_load_async_to_lds_b128 %0, %1, off"
               :: "v"(lds_off), "v"(gptr) : "memory");
}
__device__ __forceinline__ void wait_async0() {
#if __has_builtin(__builtin_amdgcn_s_wait_asynccnt)
  __builtin_amdgcn_s_wait_asynccnt(0);
#else
  asm volatile("s_wait_asynccnt 0x0" ::: "memory");
#endif
}

// A-fragment (16x32 bf16, row-major source, stride ld halves).
// lanes 0-15 -> M=lane, K {0..7,16..23}; lanes 16-31 -> K {8..15,24..31}
__device__ __forceinline__ v16bf frag_a_ld(const bf16* p, int ld) {
  int lane = threadIdx.x & 31;
  int m = lane & 15;
  int kh = lane >> 4;
  const bf16* q = p + m * ld + kh * 8;
  v16bf r;
  *(v8bf*)&r       = *(const v8bf*)(q);
  *((v8bf*)&r + 1) = *(const v8bf*)(q + 16);
  return r;
}

// B-fragment (32x16 bf16) from an N-major (transposed) source: row n holds K
// contiguously with stride ld. Lane = N-col; lanes 0-15 K {0..7,16..23}, lanes 16-31 K {8..15,24..31}.
__device__ __forceinline__ v16bf frag_b_ld(const bf16* p, int ld) {
  int lane = threadIdx.x & 31;
  int n  = lane & 15;
  int k0 = (lane >> 4) * 8;
  const bf16* q = p + n * ld + k0;
  v16bf r;
  *(v8bf*)&r       = *(const v8bf*)(q);
  *((v8bf*)&r + 1) = *(const v8bf*)(q + 16);
  return r;
}

// ---------------------------------------------------------------------------
// Tiled transpose + f32->bf16 convert:  Wt[n*K + k] = W[k*N + n]
// (coalesced both directions via 32x32 LDS tile; K,N multiples of 32)
// ---------------------------------------------------------------------------
__global__ __launch_bounds__(256) void wt_convert_kernel(const float* __restrict__ w,
                                                         bf16* __restrict__ wt,
                                                         int K, int N) {
  __shared__ float tile[32][33];
  const int kb = blockIdx.y * 32, nb = blockIdx.x * 32;
  const int tx = threadIdx.x & 31, ty = threadIdx.x >> 5;  // 8 row-groups
#pragma unroll
  for (int i = ty; i < 32; i += 8)
    tile[i][tx] = w[(size_t)(kb + i) * N + nb + tx];
  __syncthreads();
#pragma unroll
  for (int i = ty; i < 32; i += 8)
    wt[(size_t)(nb + i) * K + kb + tx] = (bf16)tile[tx][i];
}

__global__ void cvt_copy_kernel(const float* __restrict__ in, float* __restrict__ of,
                                bf16* __restrict__ ob, int n) {
  int i = blockIdx.x * 256 + threadIdx.x;
  if (i < n) { float v = in[i]; of[i] = v; ob[i] = (bf16)v; }
}

__global__ void bias_concat_kernel(const float* __restrict__ a, const float* __restrict__ b,
                                   const float* __restrict__ c, float* __restrict__ o) {
  int i = blockIdx.x * 256 + threadIdx.x;
  if (i < H_)            o[i] = a[i];
  else if (i < 2 * H_)   o[i] = b[i - H_];
  else if (i < 3 * H_)   o[i] = c[i - 2 * H_];
}

// ---------------------------------------------------------------------------
// Tiled bf16 WMMA GEMM: C[M,N] = A[M,K] * Bt[N,K]^T + bias (+res, +gelu)
// 128x128x32 tile, 256 thr = 8 waves, wave tile 32x64 (2x4 frags).
// Double-buffered LDS fed by global_load_async_to_lds_b128 (ASYNCcnt).
// LAYOUT: 0 row-major bf16; 3 fused-QKV scatter (Q/K [b,h,s,d], V [b,h,d,s]).
// ---------------------------------------------------------------------------
#define LDS_STRIDE 40  // 32 + 8 pad; keeps 16B alignment, spreads banks

template<bool RES, bool GELU_, bool SF32, bool SB16, int LAYOUT>
__global__ __launch_bounds__(256) void gemm_bf16_kernel(
    const bf16* __restrict__ A, const bf16* __restrict__ Bt,
    const float* __restrict__ bias, const float* __restrict__ residual,
    float* __restrict__ outF, bf16* __restrict__ outB,
    bf16* __restrict__ outB2, bf16* __restrict__ outB3,
    int M, int N, int K) {
  __shared__ __align__(16) bf16 As[2][128 * LDS_STRIDE];
  __shared__ __align__(16) bf16 Bs[2][128 * LDS_STRIDE];

  const int tid = threadIdx.x;
  const int wid = tid >> 5;
  const int wm = wid & 3;      // 4 wave rows
  const int wn = wid >> 2;     // 2 wave cols
  const int rowBase = blockIdx.y * 128;
  const int colBase = blockIdx.x * 128;

  // staging coordinates: this thread owns rows r0 and r0+64, one 16B chunk each
  const int r0  = tid >> 2;
  const int kc0 = (tid & 3) * 8;
  const bf16* Ap0 = A  + (size_t)(rowBase + r0)      * K + kc0;
  const bf16* Ap1 = A  + (size_t)(rowBase + r0 + 64) * K + kc0;
  const bf16* Bp0 = Bt + (size_t)(colBase + r0)      * K + kc0;
  const bf16* Bp1 = Bt + (size_t)(colBase + r0 + 64) * K + kc0;
  unsigned ao0[2], ao1[2], bo0[2], bo1[2];
#pragma unroll
  for (int bI = 0; bI < 2; ++bI) {
    ao0[bI] = (unsigned)(size_t)&As[bI][r0 * LDS_STRIDE + kc0];
    ao1[bI] = (unsigned)(size_t)&As[bI][(r0 + 64) * LDS_STRIDE + kc0];
    bo0[bI] = (unsigned)(size_t)&Bs[bI][r0 * LDS_STRIDE + kc0];
    bo1[bI] = (unsigned)(size_t)&Bs[bI][(r0 + 64) * LDS_STRIDE + kc0];
  }
  auto stage = [&](int buf, int kel) {
    async_cp16(ao0[buf], Ap0 + kel);
    async_cp16(ao1[buf], Ap1 + kel);
    async_cp16(bo0[buf], Bp0 + kel);
    async_cp16(bo1[buf], Bp1 + kel);
  };

  v8f acc[2][4];
#pragma unroll
  for (int i = 0; i < 2; ++i)
#pragma unroll
    for (int j = 0; j < 4; ++j) acc[i][j] = (v8f){0.f,0.f,0.f,0.f,0.f,0.f,0.f,0.f};

  const int nk = K / 32;
  stage(0, 0);
  wait_async0();
  __syncthreads();

  for (int t = 0; t < nk; ++t) {
    if (t + 1 < nk) stage((t + 1) & 1, (t + 1) * 32);  // prefetch next tile async

    const bf16* as = As[t & 1];
    const bf16* bs = Bs[t & 1];
    v16bf af[2], bfr[4];
#pragma unroll
    for (int i = 0; i < 2; ++i) af[i] = frag_a_ld(as + (wm * 32 + i * 16) * LDS_STRIDE, LDS_STRIDE);
#pragma unroll
    for (int j = 0; j < 4; ++j) bfr[j] = frag_b_ld(bs + (wn * 64 + j * 16) * LDS_STRIDE, LDS_STRIDE);
#pragma unroll
    for (int i = 0; i < 2; ++i)
#pragma unroll
      for (int j = 0; j < 4; ++j) acc[i][j] = wmma_bf16(af[i], bfr[j], acc[i][j]);

    wait_async0();     // next tile landed in LDS
    __syncthreads();   // everyone done reading current tile
  }

  // epilogue; C/D layout: VGPR r -> M = r + 8*(lane>=16), N = lane&15
  const int lane = tid & 31;
  const int ln  = lane & 15;
  const int lm8 = (lane >> 4) * 8;
#pragma unroll
  for (int i = 0; i < 2; ++i) {
#pragma unroll
    for (int j = 0; j < 4; ++j) {
      int col = colBase + wn * 64 + j * 16 + ln;
      float bi = bias[col];
#pragma unroll
      for (int r = 0; r < 8; ++r) {
        int row = rowBase + wm * 32 + i * 16 + lm8 + r;
        float v = acc[i][j][r] + bi;
        if (RES)   v += residual[(size_t)row * N + col];
        if (GELU_) v = 0.5f * v * (1.0f + erff(v * 0.70710678118654752f));
        if (SF32)  outF[(size_t)row * N + col] = v;
        if (SB16) {
          if (LAYOUT == 0) {
            outB[(size_t)row * N + col] = (bf16)v;     // row-major [tok, chan]
          } else {                                     // fused QKV scatter
            int seg = col / H_;
            int c   = col - seg * H_;
            int b = row >> 10, s = row & 1023;         // S_ = 1024
            int hh = c >> 6,  d = c & 63;              // HD_ = 64
            if (seg == 0)
              outB [(((size_t)(b * NH_ + hh)) * S_ + s) * HD_ + d] = (bf16)v;  // Q [b,h,s,d]
            else if (seg == 1)
              outB2[(((size_t)(b * NH_ + hh)) * S_ + s) * HD_ + d] = (bf16)v;  // K [b,h,s,d]
            else
              outB3[(((size_t)(b * NH_ + hh)) * HD_ + d) * S_ + s] = (bf16)v;  // V [b,h,d,s]
          }
        }
      }
    }
  }
}

// ---------------------------------------------------------------------------
// Attention: one 16-row Q tile per block (128 thr = 4 waves).
// scores (WMMA) -> scale+mask -> softmax (LDS) -> ctx = P @ V (WMMA)
// ---------------------------------------------------------------------------
__global__ __launch_bounds__(128) void attention_kernel(
    const bf16* __restrict__ qh, const bf16* __restrict__ kh,
    const bf16* __restrict__ vt, const float* __restrict__ mask,
    bf16* __restrict__ ctx) {
  __shared__ __align__(16) bf16 Qs[16 * HD_];       // 2 KB
  __shared__ __align__(16) bf16 Ps[16 * S_];        // 32 KB: scores then probs
  __shared__ float red[16][8];
  __shared__ float rowmax[16], rowinv[16];

  const int blk = blockIdx.x;
  const int qt = blk & 63;           // S_/16
  const int bh = blk >> 6;           // b*NH + h
  const int b = bh / NH_;
  const int h = bh % NH_;
  const int tid = threadIdx.x;
  const int wid = tid >> 5;
  const int lane = tid & 31;
  const int ln  = lane & 15;
  const int lm8 = (lane >> 4) * 8;
  const float* mrow = mask + b * S_;

  {
    const bf16* src = qh + ((size_t)bh * S_ + qt * 16) * HD_;
    *(v8bf*)(Qs + tid * 8) = *(const v8bf*)(src + tid * 8);
  }
  __syncthreads();

  v16bf aq0 = frag_a_ld(Qs, HD_);
  v16bf aq1 = frag_a_ld(Qs + 32, HD_);

  // ---- scores: wave w covers key tiles [w*16, w*16+16) ----
  for (int kt = wid * 16; kt < wid * 16 + 16; ++kt) {
    int ks = kt * 16;
    const bf16* kb = kh + ((size_t)bh * S_ + ks) * HD_;
    v16bf b0 = frag_b_ld(kb, HD_);
    v16bf b1 = frag_b_ld(kb + 32, HD_);
    v8f sc = (v8f){0.f,0.f,0.f,0.f,0.f,0.f,0.f,0.f};
    sc = wmma_bf16(aq0, b0, sc);
    sc = wmma_bf16(aq1, b1, sc);
    float mv = mrow[ks + ln];
#pragma unroll
    for (int r = 0; r < 8; ++r)
      Ps[(lm8 + r) * S_ + ks + ln] = (bf16)(sc[r] * 0.125f + mv);  // 1/sqrt(64)
  }
  __syncthreads();

  // ---- softmax over 1024: 8 threads per row x 128 cols ----
  {
    const int row = tid >> 3, seg = tid & 7;
    bf16* prow = Ps + row * S_ + seg * 128;
    float mx = -3.0e38f;
    for (int c = 0; c < 128; ++c) mx = fmaxf(mx, (float)prow[c]);
    red[row][seg] = mx;
    __syncthreads();
    if (seg == 0) {
      float m = red[row][0];
      for (int i = 1; i < 8; ++i) m = fmaxf(m, red[row][i]);
      rowmax[row] = m;
    }
    __syncthreads();
    float rm = rowmax[row];
    float sum = 0.f;
    for (int c = 0; c < 128; ++c) {
      float e = __expf((float)prow[c] - rm);
      sum += e;
      prow[c] = (bf16)e;
    }
    red[row][seg] = sum;
    __syncthreads();
    if (seg == 0) {
      float s = 0.f;
      for (int i = 0; i < 8; ++i) s += red[row][i];
      rowinv[row] = 1.0f / s;
    }
  }
  __syncthreads();

  // ---- ctx = P @ V : wave w owns hd cols [w*16, w*16+16) ----
  v8f cacc = (v8f){0.f,0.f,0.f,0.f,0.f,0.f,0.f,0.f};
  const bf16* vbase = vt + ((size_t)bh * HD_ + wid * 16) * S_;
  for (int k0 = 0; k0 < S_; k0 += 32) {
    v16bf ap = frag_a_ld(Ps + k0, S_);
    v16bf bv = frag_b_ld(vbase + k0, S_);
    cacc = wmma_bf16(ap, bv, cacc);
  }
#pragma unroll
  for (int r = 0; r < 8; ++r) {
    int rr = lm8 + r;
    float v = cacc[r] * rowinv[rr];
    int s = qt * 16 + rr;
    ctx[((size_t)(b * S_ + s)) * H_ + h * HD_ + wid * 16 + ln] = (bf16)v;
  }
}

// ---------------------------------------------------------------------------
// LayerNorm over H=768 per token; emits fp32 + bf16
// ---------------------------------------------------------------------------
__global__ __launch_bounds__(256) void layernorm_kernel(
    const float* __restrict__ in, const float* __restrict__ g,
    const float* __restrict__ be, float* __restrict__ outF, bf16* __restrict__ outB) {
  __shared__ float rs[8], rq[8];
  __shared__ float mean_s, rstd_s;
  const int t = threadIdx.x;
  const float* x = in + (size_t)blockIdx.x * H_;
  float v0 = x[t], v1 = x[t + 256], v2 = x[t + 512];
  float s = v0 + v1 + v2;
  float q = v0 * v0 + v1 * v1 + v2 * v2;
#pragma unroll
  for (int o = 16; o > 0; o >>= 1) {
    s += __shfl_xor(s, o, 32);
    q += __shfl_xor(q, o, 32);
  }
  if ((t & 31) == 0) { rs[t >> 5] = s; rq[t >> 5] = q; }
  __syncthreads();
  if (t == 0) {
    float S = 0.f, Q = 0.f;
    for (int i = 0; i < 8; ++i) { S += rs[i]; Q += rq[i]; }
    float m = S * (1.0f / H_);
    float var = Q * (1.0f / H_) - m * m;
    mean_s = m;
    rstd_s = rsqrtf(var + 1e-5f);
  }
  __syncthreads();
  float m = mean_s, rstd = rstd_s;
  size_t base = (size_t)blockIdx.x * H_;
#pragma unroll
  for (int i = 0; i < 3; ++i) {
    int c = t + i * 256;
    float xv = (i == 0) ? v0 : (i == 1) ? v1 : v2;
    float y = (xv - m) * rstd * g[c] + be[c];
    outF[base + c] = y;
    outB[base + c] = (bf16)y;
  }
}

// ---------------------------------------------------------------------------
// Host driver
// ---------------------------------------------------------------------------
extern "C" void kernel_launch(void* const* d_in, const int* in_sizes, int n_in,
                              void* d_out, int out_size, void* d_ws, size_t ws_size,
                              hipStream_t stream) {
  const float* x_in  = (const float*)d_in[0];
  const float* mask  = (const float*)d_in[1];
  const float* q_w   = (const float*)d_in[2];
  const float* q_b   = (const float*)d_in[3];
  const float* k_w   = (const float*)d_in[4];
  const float* k_b   = (const float*)d_in[5];
  const float* v_w   = (const float*)d_in[6];
  const float* v_b   = (const float*)d_in[7];
  const float* ao_w  = (const float*)d_in[8];
  const float* ao_b  = (const float*)d_in[9];
  const float* ln1_g = (const float*)d_in[10];
  const float* ln1_b = (const float*)d_in[11];
  const float* ff1_w = (const float*)d_in[12];
  const float* ff1_b = (const float*)d_in[13];
  const float* ff2_w = (const float*)d_in[14];
  const float* ff2_b = (const float*)d_in[15];
  const float* ln2_g = (const float*)d_in[16];
  const float* ln2_b = (const float*)d_in[17];

  char* ws = (char*)d_ws;
  auto alloc = [&](size_t bytes) -> char* {
    char* p = ws;
    ws += (bytes + 255) & ~(size_t)255;
    return p;
  };

  bf16*  qkvT  = (bf16*)alloc((size_t)NQKV * H_ * 2);  // [2304][768]
  float* qkvB  = (float*)alloc((size_t)NQKV * 4);
  bf16*  aowT  = (bf16*)alloc((size_t)H_ * H_ * 2);
  bf16*  ff1T  = (bf16*)alloc((size_t)H_ * FF_ * 2);
  bf16*  ff2T  = (bf16*)alloc((size_t)H_ * FF_ * 2);
  float* xF    = (float*)alloc((size_t)MTOK * H_ * 4);
  bf16*  xB    = (bf16*)alloc((size_t)MTOK * H_ * 2);
  bf16*  qB    = (bf16*)alloc((size_t)MTOK * H_ * 2);
  bf16*  kB    = (bf16*)alloc((size_t)MTOK * H_ * 2);
  bf16*  vB    = (bf16*)alloc((size_t)MTOK * H_ * 2);
  bf16*  ctxB  = (bf16*)alloc((size_t)MTOK * H_ * 2);
  float* attnF = (float*)alloc((size_t)MTOK * H_ * 4);
  bf16*  attnB = (bf16*)alloc((size_t)MTOK * H_ * 2);
  float* tmpF  = (float*)alloc((size_t)MTOK * H_ * 4);
  bf16*  hB    = (bf16*)alloc((size_t)MTOK * FF_ * 2);

  // --- one-time conversions (tiled, coalesced) ---
  dim3 tHH(H_ / 32, H_ / 32);
  dim3 tHF(FF_ / 32, H_ / 32);   // (N/32, K/32) for W[K][N]
  dim3 tFH(H_ / 32, FF_ / 32);
  wt_convert_kernel<<<tHH, 256, 0, stream>>>(q_w,  qkvT,                        H_, H_);
  wt_convert_kernel<<<tHH, 256, 0, stream>>>(k_w,  qkvT + (size_t)H_ * H_,      H_, H_);
  wt_convert_kernel<<<tHH, 256, 0, stream>>>(v_w,  qkvT + (size_t)2 * H_ * H_,  H_, H_);
  wt_convert_kernel<<<tHH, 256, 0, stream>>>(ao_w, aowT, H_, H_);
  wt_convert_kernel<<<tHF, 256, 0, stream>>>(ff1_w, ff1T, H_, FF_);
  wt_convert_kernel<<<tFH, 256, 0, stream>>>(ff2_w, ff2T, FF_, H_);
  bias_concat_kernel<<<(NQKV + 255) / 256, 256, 0, stream>>>(q_b, k_b, v_b, qkvB);
  cvt_copy_kernel<<<(MTOK * H_ + 255) / 256, 256, 0, stream>>>(x_in, xF, xB, MTOK * H_);

  dim3 gH(H_ / 128, MTOK / 128);      // N=768
  dim3 gQKV(NQKV / 128, MTOK / 128);  // N=2304
  dim3 gF(FF_ / 128, MTOK / 128);     // N=3072

  for (int layer = 0; layer < 6; ++layer) {
    // fused QKV: one pass over x, scatter to per-head layouts
    gemm_bf16_kernel<false, false, false, true, 3><<<gQKV, 256, 0, stream>>>(
        xB, qkvT, qkvB, nullptr, nullptr, qB, kB, vB, MTOK, NQKV, H_);

    attention_kernel<<<B_ * NH_ * (S_ / 16), 128, 0, stream>>>(qB, kB, vB, mask, ctxB);

    // self-output: ctx@ao_w + ao_b + x  (fp32) -> LN1
    gemm_bf16_kernel<true, false, true, false, 0><<<gH, 256, 0, stream>>>(
        ctxB, aowT, ao_b, xF, tmpF, nullptr, nullptr, nullptr, MTOK, H_, H_);
    layernorm_kernel<<<MTOK, 256, 0, stream>>>(tmpF, ln1_g, ln1_b, attnF, attnB);

    // FFN: GELU(attn@ff1 + b) -> @ff2 + b + attn -> LN2
    gemm_bf16_kernel<false, true, false, true, 0><<<gF, 256, 0, stream>>>(
        attnB, ff1T, ff1_b, nullptr, nullptr, hB, nullptr, nullptr, MTOK, FF_, H_);
    gemm_bf16_kernel<true, false, true, false, 0><<<gH, 256, 0, stream>>>(
        hB, ff2T, ff2_b, attnF, tmpF, nullptr, nullptr, nullptr, MTOK, H_, FF_);

    float* outPtr = (layer == 5) ? (float*)d_out : xF;
    layernorm_kernel<<<MTOK, 256, 0, stream>>>(tmpF, ln2_g, ln2_b, outPtr, xB);
  }
}
